// GraphRes_1236950582171
// MI455X (gfx1250) — matrix operand order repaired and checked by reading, not compiled, not run on hardware
//
#include <hip/hip_runtime.h>
#include <math.h>

// ---------------------------------------------------------------------------
// GraphRes (SplineConv GNN) for MI455X / gfx1250, wave32, WMMA f16->f32.
// Per-edge spline GEMMs on v_wmma_f32_16x16x32_f16: 16-edge M tiles x
// 16-channel N tiles, K unrolled in 32-chunks (templated), A fragments built
// register-resident per lane, B fragments pre-permuted to WMMA layout and
// staged through LDS as full v16h (ds_load_b128). Segment-sum scatter via
// global_atomic_add_f32.
// ---------------------------------------------------------------------------

typedef __attribute__((ext_vector_type(16))) _Float16 v16h;
typedef __attribute__((ext_vector_type(8)))  float    v8f;

#define N_NODES 8192
#define N_EDGES 131072
#define BATCHSZ 4
#define NPB     (N_NODES / BATCHSZ)   // 2048
#define NCLUST  (BATCHSZ * 72)        // 288
#define NCELL7  (BATCHSZ * 16)        // 64

// ------------------------------- device utils ------------------------------

__device__ __forceinline__ unsigned int hash_u32(unsigned int x) {
  x ^= x >> 16; x *= 0x7feb352du;
  x ^= x >> 15; x *= 0x846ca68bu;
  x ^= x >> 16; return x;
}
__device__ __forceinline__ float u01(unsigned int h) {
  return (float)(h >> 8) * (1.0f / 16777216.0f);
}
__device__ __forceinline__ float gauss_hash(unsigned int seed) {
  float s = 0.f;
  #pragma unroll
  for (int i = 0; i < 4; ++i) s += u01(hash_u32(seed * 4u + (unsigned)i + 0x9e3779b9u));
  return (s - 2.0f) * 1.7320508f;   // var ~= 1
}
__device__ __forceinline__ float clamp01(float v) {
  return fminf(fmaxf(v, 0.f), 1.f);
}
__device__ __forceinline__ void atomicMaxF(float* addr, float val) {
  unsigned int* ua = (unsigned int*)addr;
  unsigned int old = __float_as_uint(*addr);
  while (val > __uint_as_float(old)) {
    unsigned int assumed = old;
    old = atomicCAS(ua, assumed, __float_as_uint(val));
    if (old == assumed) break;
  }
}

// ------------------------------- tiny kernels -------------------------------

__global__ void k_fillf(float* p, float v, int n) {
  int i = blockIdx.x * blockDim.x + threadIdx.x;
  if (i < n) p[i] = v;
}
__global__ void k_copy(float* d, const float* s, int n) {
  int i = blockIdx.x * blockDim.x + threadIdx.x;
  if (i < n) d[i] = s[i];
}

// basis weights per edge (kernel_size=2, degree-1 B-spline -> 8 products) + degree
__global__ void k_edge_basis(const float* __restrict__ attr,
                             const int* __restrict__ edst,
                             const int* __restrict__ cmap,
                             float* __restrict__ wE, float* __restrict__ deg, int E) {
  int e = blockIdx.x * blockDim.x + threadIdx.x;
  if (e >= E) return;
  float f0 = clamp01(attr[e * 3 + 0]);
  float f1 = clamp01(attr[e * 3 + 1]);
  float f2 = clamp01(attr[e * 3 + 2]);
  #pragma unroll
  for (int s = 0; s < 8; ++s) {
    float w = ((s & 1) ? f0 : 1.f - f0) * ((s & 2) ? f1 : 1.f - f1) * ((s & 4) ? f2 : 1.f - f2);
    wE[e * 8 + s] = w;
  }
  int d = edst[e];
  if (cmap) d = cmap[d];
  atomicAdd(&deg[d], 1.0f);
}

// W [K x Cout] f32 -> f16 in WMMA B-fragment order:
// Bf[((nt*kChunks + c32)*32 + lane)*16 + h] = W[c32*32 + (h&7) + 16*(h>>3) + 8*(lane>>4)][nt*16 + (lane&15)]
template<int Cin, int Cout>
__global__ void k_convert_w_frag(const float* __restrict__ W, _Float16* __restrict__ Bf) {
  constexpr int K = 8 * Cin;
  constexpr int Kp = (K + 31) & ~31;
  constexpr int kChunks = Kp / 32;
  constexpr int nTiles = (Cout + 15) / 16;
  int i = blockIdx.x * blockDim.x + threadIdx.x;
  if (i >= nTiles * kChunks * 32 * 16) return;
  int h = i & 15;
  int lane = (i >> 4) & 31;
  int c32 = (i >> 9) % kChunks;
  int nt = (i >> 9) / kChunks;
  int k = c32 * 32 + (h & 7) + ((h >> 3) << 4) + ((lane >> 4) << 3);
  int n = nt * 16 + (lane & 15);
  float v = (k < K && n < Cout) ? W[k * Cout + n] : 0.f;
  Bf[i] = (_Float16)v;
}

// build A fragments for one lane (row m fixed); KOFF = 0 (lanes 0-15) or 8 (16-31).
// All indices constant after unroll -> pure register indexing.
template<int Cin, int KOFF, int kChunks>
__device__ __forceinline__ void build_afrag(const float* wer, const float* xr,
                                            v16h* afrag) {
  constexpr int K = 8 * Cin;
  #pragma unroll
  for (int c32 = 0; c32 < kChunks; ++c32) {
    #pragma unroll
    for (int h = 0; h < 16; ++h) {
      const int k = c32 * 32 + (h & 7) + ((h >> 3) << 4) + KOFF;  // A 16-bit 16x32 layout
      float v = 0.f;
      if (k < K) v = wer[k / Cin] * xr[k % Cin];
      afrag[c32][h] = (_Float16)v;
    }
  }
}

// ------------------------- WMMA spline-conv GEMM ----------------------------
// Wave = one 16-edge tile. A[m][j] = wE[e_m][j/Cin] * x[src(e_m)][j%Cin],
// B[j][n] = W[j][n]; D scattered with atomic adds to acc[dst(e_m)][n].
template<int Cin, int Cout, bool UseCmap>
__global__ __launch_bounds__(128) void k_spline_gemm_t(
    const float* __restrict__ xin, const float* __restrict__ wE,
    const int* __restrict__ esrc, const int* __restrict__ edst,
    const int* __restrict__ cmap, const v16h* __restrict__ Bfrag,
    float* __restrict__ acc, int E) {
  constexpr int K = 8 * Cin;
  constexpr int Kp = (K + 31) & ~31;
  constexpr int kChunks = Kp / 32;
  constexpr int nTiles = (Cout + 15) / 16;

  __shared__ v16h sB[nTiles * kChunks * 32];   // <= 16 KB, WMMA fragment order

  const int tid = threadIdx.x;
  for (int i = tid; i < nTiles * kChunks * 32; i += 128) sB[i] = Bfrag[i];
  __syncthreads();

  const int wv = tid >> 5, lane = tid & 31;
  const int half = lane >> 4;        // 0: lanes 0-15, 1: lanes 16-31
  const int l15 = lane & 15;
  const int tileBase = (blockIdx.x * 4 + wv) * 16;
  if (tileBase >= E) return;

  __builtin_prefetch(&esrc[tileBase], 0, 1);                // global_prefetch_b8
  __builtin_prefetch(&wE[(size_t)tileBase * 8], 0, 1);

  // this lane's A-row: edge m = l15
  const int e = tileBase + l15;
  int s = esrc[e];
  if (UseCmap) s = cmap[s];

  float wer[8];
  #pragma unroll
  for (int kk = 0; kk < 8; ++kk) wer[kk] = wE[e * 8 + kk];
  float xr[Cin];
  #pragma unroll
  for (int ci = 0; ci < Cin; ++ci) xr[ci] = xin[s * Cin + ci];

  v16h afrag[kChunks];
  if (half == 0) build_afrag<Cin, 0, kChunks>(wer, xr, afrag);
  else           build_afrag<Cin, 8, kChunks>(wer, xr, afrag);

  #pragma unroll
  for (int nt = 0; nt < nTiles; ++nt) {
    v8f c = {};
    #pragma unroll
    for (int c32 = 0; c32 < kChunks; ++c32) {
      v16h b = sB[(nt * kChunks + c32) * 32 + lane];        // ds_load_b128 x2
      c = __builtin_amdgcn_wmma_f32_16x16x32_f16(false, afrag[c32], false, b,
                                                 (short)0, c, false, false);
    }
    const int n = nt * 16 + l15;
    if (n < Cout) {
      #pragma unroll
      for (int v = 0; v < 8; ++v) {
        const int m = v + (half << 3);        // C/D layout: m = vgpr + 8*half
        int d = edst[tileBase + m];
        if (UseCmap) d = cmap[d];
        atomicAdd(&acc[d * Cout + n], c[v]);  // segment_sum over dst
      }
    }
  }
}

// mean-aggregate + ELU + batchnorm stats
__global__ void k_post1(float* acc, const float* deg, float* bnsum, float* bnsq,
                        int n, int Cout) {
  int i = blockIdx.x * blockDim.x + threadIdx.x;
  if (i >= n * Cout) return;
  int node = i / Cout, c = i - node * Cout;
  float d = fmaxf(deg[node], 1.0f);
  float v = acc[i] / d;
  v = v > 0.f ? v : (expf(v) - 1.f);          // ELU(alpha=1)
  acc[i] = v;
  atomicAdd(&bnsum[c], v);
  atomicAdd(&bnsq[c], v * v);
}

// batchnorm apply (+ optional residual)
__global__ void k_post2(const float* acc, const float* bnsum, const float* bnsq,
                        const float* g, const float* b, float* out,
                        const float* resid, int n, int Cout) {
  int i = blockIdx.x * blockDim.x + threadIdx.x;
  if (i >= n * Cout) return;
  int c = i % Cout;
  float inv_n = 1.0f / (float)n;
  float mu = bnsum[c] * inv_n;
  float var = bnsq[c] * inv_n - mu * mu;
  float y = (acc[i] - mu) * rsqrtf(var + 1e-5f) * g[c] + b[c];
  if (resid) y += resid[i];
  out[i] = y;
}

// ------------------------------ tmd_map (4x4 L) -----------------------------

__global__ void k_scatter_adj(const int* __restrict__ esrc, const int* __restrict__ edst,
                              const float* __restrict__ r, float* __restrict__ tmp, int E) {
  int e = blockIdx.x * blockDim.x + threadIdx.x;
  if (e >= E) return;
  atomicAdd(&tmp[esrc[e]], r[edst[e]]);       // (adj @ r)[src] += r[dst]
}
__global__ void k_raccum(float* r, const float* tmp, float* red, int n) {
  int i = blockIdx.x * blockDim.x + threadIdx.x;
  if (i >= n) return;
  float v = r[i] + tmp[i];
  r[i] = v;
  atomicAdd(&red[0], v);
  atomicAdd(&red[1], v * v);
}
__global__ void k_rff(const float* __restrict__ r, const float* __restrict__ red,
                      float* sgn, float* ksum, int iter, int n) {
  __shared__ float sdot[4][256];
  float mu = red[0] / (float)n;
  float var = red[1] / (float)n - mu * mu;
  float rstd = rsqrtf(fmaxf(var, 1e-12f));
  int t = threadIdx.x;
  int rows = n / BATCHSZ;
  float accb[4] = {0.f, 0.f, 0.f, 0.f};
  for (int j = t; j < rows; j += 256) {
    float wj = gauss_hash((unsigned)(iter * 3 + 0) * 0x10001u + (unsigned)j);
    #pragma unroll
    for (int b = 0; b < 4; ++b) accb[b] += ((r[b * rows + j] - mu) * rstd) * wj;
  }
  #pragma unroll
  for (int b = 0; b < 4; ++b) sdot[b][t] = accb[b];
  __syncthreads();
  for (int s = 128; s > 0; s >>= 1) {
    if (t < s) {
      #pragma unroll
      for (int b = 0; b < 4; ++b) sdot[b][t] += sdot[b][t + s];
    }
    __syncthreads();
  }
  if (t < 4) {
    float bb = gauss_hash((unsigned)(iter * 3 + 1) * 0x20003u + (unsigned)t) * 6.2831853f;
    float q = cosf(sdot[t][0] + bb);
    float tv = u01(hash_u32((unsigned)(iter * 3 + 2) * 0x40005u + (unsigned)t)) * 2.f - 1.f;
    sgn[t] = (q + tv < 0.f) ? 0.f : 1.f;
  }
  __syncthreads();
  if (t < 16) {
    int p = t >> 2, q = t & 3;
    ksum[t] += (sgn[p] != sgn[q]) ? 1.f : 0.f;
  }
}
__global__ void k_make_L(const float* ksum, float* L) {
  if (threadIdx.x != 0 || blockIdx.x != 0) return;
  float ke[16], qt[4], dt_[4];
  for (int i = 0; i < 16; ++i) ke[i] = ksum[i] / 4.f;
  for (int j = 0; j < 4; ++j) {
    float s = 0.f;
    for (int i = 0; i < 4; ++i) s += ke[j * 4 + i];    // row sums (axis=1)
    qt[j] = fmaxf(s, 1e-6f);
  }
  float kt[16];
  for (int i = 0; i < 4; ++i)
    for (int j = 0; j < 4; ++j) kt[i * 4 + j] = ke[i * 4 + j] * (0.25f / qt[j]);
  for (int i = 0; i < 4; ++i) {
    float s = 0.f;
    for (int j = 0; j < 4; ++j) s += kt[i * 4 + j];
    dt_[i] = fmaxf(s, 1e-6f);
  }
  for (int i = 0; i < 4; ++i)
    for (int j = 0; j < 4; ++j)
      L[i * 4 + j] = 4.0f * (kt[i * 4 + j] / dt_[i]) - (i == j ? 1.f : 0.f);
}
__global__ void k_tmd_res(const float* __restrict__ h, const float* __restrict__ L,
                          const float* __restrict__ dt, float* __restrict__ out) {
  int i = blockIdx.x * blockDim.x + threadIdx.x;
  if (i >= N_NODES * 32) return;
  int node = i >> 5, c = i & 31;
  int b = node / NPB, il = node - b * NPB;
  float s = 0.f;
  #pragma unroll
  for (int b2 = 0; b2 < 4; ++b2) s += L[b * 4 + b2] * h[(b2 * NPB + il) * 32 + c];
  out[i] = h[i] + dt[0] * s;
}

// --------------------------------- pooling ----------------------------------

__global__ void k_pool5(const float* __restrict__ h, const float* __restrict__ pos,
                        const int* __restrict__ batch, int* __restrict__ cluster,
                        float* __restrict__ hp, float* __restrict__ cnt,
                        float* __restrict__ posp) {
  int i = blockIdx.x * blockDim.x + threadIdx.x;
  if (i >= N_NODES) return;
  float px = pos[i * 3 + 0], py = pos[i * 3 + 1], pz = pos[i * 3 + 2];
  int cx = min(max((int)floorf(px / 16.f), 0), 7);
  int cy = min(max((int)floorf(py / 12.f), 0), 8);
  int cl = batch[i] * 72 + cx * 9 + cy;
  cluster[i] = cl;
  atomicAdd(&cnt[cl], 1.f);
  atomicAdd(&posp[cl * 3 + 0], px);
  atomicAdd(&posp[cl * 3 + 1], py);
  atomicAdd(&posp[cl * 3 + 2], pz);
  for (int c = 0; c < 32; ++c) atomicMaxF(&hp[cl * 32 + c], h[i * 32 + c]);
}
__global__ void k_pool5_fix(float* hp, float* posp, const float* cnt, int nc) {
  int i = blockIdx.x * blockDim.x + threadIdx.x;
  if (i < nc * 32) {
    float v = hp[i];
    if (!isfinite(v)) hp[i] = 0.f;
  }
  if (i < nc * 3) {
    int cl = i / 3;
    posp[i] = posp[i] / fmaxf(cnt[cl], 1.f);
  }
}
__global__ void k_cart(const int* __restrict__ esrc, const int* __restrict__ edst,
                       const int* __restrict__ cluster, const float* __restrict__ posp,
                       float* __restrict__ cart, float* __restrict__ maxabs, int E) {
  int e = blockIdx.x * blockDim.x + threadIdx.x;
  if (e >= E) return;
  int cs = cluster[esrc[e]], cd = cluster[edst[e]];
  float mloc = 0.f;
  #pragma unroll
  for (int d = 0; d < 3; ++d) {
    float v = posp[cd * 3 + d] - posp[cs * 3 + d];
    cart[e * 3 + d] = v;
    mloc = fmaxf(mloc, fabsf(v));
  }
  atomicMax((unsigned int*)maxabs, __float_as_uint(mloc));  // non-negative floats
}
__global__ void k_attr_norm(float* attr, const float* maxabs, int n) {
  int i = blockIdx.x * blockDim.x + threadIdx.x;
  if (i >= n) return;
  attr[i] = attr[i] / (2.f * fmaxf(maxabs[0], 1e-9f)) + 0.5f;
}
__global__ void k_cl7(const float* __restrict__ posp, int* __restrict__ cl7) {
  int c = blockIdx.x * blockDim.x + threadIdx.x;
  if (c >= NCLUST) return;
  int px = min(max((int)floorf(posp[c * 3 + 0] / 30.f), 0), 3);
  int py = min(max((int)floorf(posp[c * 3 + 1] / 25.f), 0), 3);
  cl7[c] = (c / 72) * 16 + px * 4 + py;
}
__global__ void k_pool7(const float* __restrict__ hp, const int* __restrict__ cl7,
                        float* __restrict__ hx) {
  int c = blockIdx.x * blockDim.x + threadIdx.x;
  if (c >= NCLUST) return;
  int cl = cl7[c];
  for (int ch = 0; ch < 32; ++ch) atomicMaxF(&hx[cl * 32 + ch], hp[c * 32 + ch]);
}
__global__ void k_hx_fix(float* hx, int n) {
  int i = blockIdx.x * blockDim.x + threadIdx.x;
  if (i >= n) return;
  if (!isfinite(hx[i])) hx[i] = 0.f;
}
__global__ void k_fc(const float* __restrict__ hx, const float* __restrict__ wfc,
                     float* __restrict__ out) {
  int t = threadIdx.x;
  if (t >= 8) return;
  int b = t >> 1, o = t & 1;
  float s = 0.f;
  for (int k = 0; k < 512; ++k) s += hx[b * 512 + k] * wfc[k * 2 + o];
  out[b * 2 + o] = s;
}

// --------------------------------- host side --------------------------------

static inline int ngrid(int n, int b) { return (n + b - 1) / b; }

extern "C" void kernel_launch(void* const* d_in, const int* in_sizes, int n_in,
                              void* d_out, int out_size, void* d_ws, size_t ws_size,
                              hipStream_t stream) {
  (void)in_sizes; (void)n_in; (void)out_size; (void)ws_size;

  // input order: x, pos, edge_attr, {w,g,b}x7, w_fc, dt, edge_index, batch
  const float* x    = (const float*)d_in[0];
  const float* pos  = (const float*)d_in[1];
  const float* eatt = (const float*)d_in[2];
  const float* Wl[7], *gl[7], *bl[7];
  for (int l = 0; l < 7; ++l) {
    Wl[l] = (const float*)d_in[3 + 3 * l];
    gl[l] = (const float*)d_in[4 + 3 * l];
    bl[l] = (const float*)d_in[5 + 3 * l];
  }
  const float* wfc = (const float*)d_in[24];
  const float* dt  = (const float*)d_in[25];
  const int* ei    = (const int*)d_in[26];
  const int* batch = (const int*)d_in[27];
  const int* esrc = ei;
  const int* edst = ei + N_EDGES;
  float* out = (float*)d_out;

  // workspace layout (floats; Wh offset kept 32B-aligned for v16h access)
  float* W = (float*)d_ws;
  size_t o = 0;
  float* wE    = W + o; o += (size_t)N_EDGES * 8;
  float* deg   = W + o; o += N_NODES;
  float* degp  = W + o; o += NCLUST;
  float* acc   = W + o; o += (size_t)N_NODES * 32;
  float* hA    = W + o; o += (size_t)N_NODES * 32;
  float* hB    = W + o; o += (size_t)N_NODES * 32;
  float* hsc   = W + o; o += (size_t)N_NODES * 32;
  o = (o + 7) & ~(size_t)7;
  _Float16* Wh = (_Float16*)(W + o); o += (2 * 8 * 32 * 16) / 2;   // max 8192 halves
  float* bnsum = W + o; o += 32;
  float* bnsq  = W + o; o += 32;
  float* rvec  = W + o; o += N_NODES;
  float* tmp   = W + o; o += N_NODES;
  float* red   = W + o; o += 4;
  float* sgn   = W + o; o += 4;
  float* ksum  = W + o; o += 16;
  float* Lmat  = W + o; o += 16;
  float* hp    = W + o; o += (size_t)NCLUST * 32;
  float* hp2   = W + o; o += (size_t)NCLUST * 32;
  float* hpsc  = W + o; o += (size_t)NCLUST * 32;
  float* cnt   = W + o; o += NCLUST;
  float* posp  = W + o; o += NCLUST * 3;
  int* cluster = (int*)(W + o); o += N_NODES;
  int* cl7     = (int*)(W + o); o += NCLUST;
  float* cart  = W + o; o += (size_t)N_EDGES * 3;
  float* maxab = W + o; o += 1;
  float* hx    = W + o; o += (size_t)NCELL7 * 32;

  const int TB = 256;
  const dim3 gG(N_EDGES / 64), bG(128);

  // per-layer launch helpers (template dispatch)
  auto conv_w = [&](int l) {
    switch (l) {
      case 0: hipLaunchKernelGGL((k_convert_w_frag<1, 8>),   dim3(ngrid(512, TB)),  dim3(TB), 0, stream, Wl[0], Wh); break;
      case 1: hipLaunchKernelGGL((k_convert_w_frag<8, 16>),  dim3(ngrid(1024, TB)), dim3(TB), 0, stream, Wl[1], Wh); break;
      case 2: hipLaunchKernelGGL((k_convert_w_frag<16, 16>), dim3(ngrid(2048, TB)), dim3(TB), 0, stream, Wl[2], Wh); break;
      case 3: hipLaunchKernelGGL((k_convert_w_frag<16, 16>), dim3(ngrid(2048, TB)), dim3(TB), 0, stream, Wl[3], Wh); break;
      case 4: hipLaunchKernelGGL((k_convert_w_frag<16, 32>), dim3(ngrid(4096, TB)), dim3(TB), 0, stream, Wl[4], Wh); break;
      case 5: hipLaunchKernelGGL((k_convert_w_frag<32, 32>), dim3(ngrid(8192, TB)), dim3(TB), 0, stream, Wl[5], Wh); break;
      case 6: hipLaunchKernelGGL((k_convert_w_frag<32, 32>), dim3(ngrid(8192, TB)), dim3(TB), 0, stream, Wl[6], Wh); break;
    }
  };
  auto gemm = [&](int l, const float* hin_, float* acc_, const int* cm) {
    const v16h* Bf = (const v16h*)Wh;
    switch (l) {
      case 0: hipLaunchKernelGGL((k_spline_gemm_t<1, 8, false>),   gG, bG, 0, stream, hin_, wE, esrc, edst, cm, Bf, acc_, N_EDGES); break;
      case 1: hipLaunchKernelGGL((k_spline_gemm_t<8, 16, false>),  gG, bG, 0, stream, hin_, wE, esrc, edst, cm, Bf, acc_, N_EDGES); break;
      case 2:
      case 3: hipLaunchKernelGGL((k_spline_gemm_t<16, 16, false>), gG, bG, 0, stream, hin_, wE, esrc, edst, cm, Bf, acc_, N_EDGES); break;
      case 4: hipLaunchKernelGGL((k_spline_gemm_t<16, 32, false>), gG, bG, 0, stream, hin_, wE, esrc, edst, cm, Bf, acc_, N_EDGES); break;
      case 5:
      case 6: hipLaunchKernelGGL((k_spline_gemm_t<32, 32, true>),  gG, bG, 0, stream, hin_, wE, esrc, edst, cm, Bf, acc_, N_EDGES); break;
    }
  };

  // ---- edge basis + node degree ----
  hipLaunchKernelGGL(k_fillf, dim3(ngrid(N_NODES, TB)), dim3(TB), 0, stream, deg, 0.f, N_NODES);
  hipLaunchKernelGGL(k_edge_basis, dim3(ngrid(N_EDGES, TB)), dim3(TB), 0, stream,
                     eatt, edst, (const int*)nullptr, wE, deg, N_EDGES);

  // ---- tmd_map: 4 diffusion steps -> 4x4 L ----
  hipLaunchKernelGGL(k_copy, dim3(ngrid(N_NODES, TB)), dim3(TB), 0, stream, rvec, x, N_NODES);
  hipLaunchKernelGGL(k_fillf, dim3(1), dim3(32), 0, stream, ksum, 0.f, 16);
  for (int it = 0; it < 4; ++it) {
    hipLaunchKernelGGL(k_fillf, dim3(ngrid(N_NODES, TB)), dim3(TB), 0, stream, tmp, 0.f, N_NODES);
    hipLaunchKernelGGL(k_fillf, dim3(1), dim3(32), 0, stream, red, 0.f, 4);
    hipLaunchKernelGGL(k_scatter_adj, dim3(ngrid(N_EDGES, TB)), dim3(TB), 0, stream,
                       esrc, edst, rvec, tmp, N_EDGES);
    hipLaunchKernelGGL(k_raccum, dim3(ngrid(N_NODES, TB)), dim3(TB), 0, stream,
                       rvec, tmp, red, N_NODES);
    hipLaunchKernelGGL(k_rff, dim3(1), dim3(256), 0, stream, rvec, red, sgn, ksum, it, N_NODES);
  }
  hipLaunchKernelGGL(k_make_L, dim3(1), dim3(32), 0, stream, ksum, Lmat);

  // ---- node-level spline conv layers 1..5 ----
  const int coutL[7] = {8, 16, 16, 16, 32, 32, 32};
  const float* hin[5]  = {x,  hA, hB, hA, hB};
  float*       hout[5] = {hA, hB, hA, hB, hA};
  const float* res5[5] = {nullptr, nullptr, nullptr, hsc, nullptr};

  for (int l = 0; l < 5; ++l) {
    int Cout = coutL[l];
    hipLaunchKernelGGL(k_fillf, dim3(ngrid(N_NODES * Cout, TB)), dim3(TB), 0, stream,
                       acc, 0.f, N_NODES * Cout);
    hipLaunchKernelGGL(k_fillf, dim3(1), dim3(64), 0, stream, bnsum, 0.f, 32);
    hipLaunchKernelGGL(k_fillf, dim3(1), dim3(64), 0, stream, bnsq, 0.f, 32);
    conv_w(l);
    gemm(l, hin[l], acc, nullptr);
    hipLaunchKernelGGL(k_post1, dim3(ngrid(N_NODES * Cout, TB)), dim3(TB), 0, stream,
                       acc, deg, bnsum, bnsq, N_NODES, Cout);
    hipLaunchKernelGGL(k_post2, dim3(ngrid(N_NODES * Cout, TB)), dim3(TB), 0, stream,
                       acc, bnsum, bnsq, gl[l], bl[l], hout[l], res5[l], N_NODES, Cout);
    if (l == 1)  // save skip connection h_sc = h after layer 2
      hipLaunchKernelGGL(k_copy, dim3(ngrid(N_NODES * 16, TB)), dim3(TB), 0, stream,
                         hsc, hB, N_NODES * 16);
  }

  // ---- TMD diffusion residual: hB = hA + dt * (L @ hA) ----
  hipLaunchKernelGGL(k_tmd_res, dim3(ngrid(N_NODES * 32, TB)), dim3(TB), 0, stream,
                     hA, Lmat, dt, hB);

  // ---- pool5: voxel max pool to 288 clusters ----
  hipLaunchKernelGGL(k_fillf, dim3(ngrid(NCLUST * 32, TB)), dim3(TB), 0, stream,
                     hp, -INFINITY, NCLUST * 32);
  hipLaunchKernelGGL(k_fillf, dim3(ngrid(NCLUST * 3, TB)), dim3(TB), 0, stream,
                     posp, 0.f, NCLUST * 3);
  hipLaunchKernelGGL(k_fillf, dim3(ngrid(NCLUST, TB)), dim3(TB), 0, stream, cnt, 0.f, NCLUST);
  hipLaunchKernelGGL(k_pool5, dim3(ngrid(N_NODES, TB)), dim3(TB), 0, stream,
                     hB, pos, batch, cluster, hp, cnt, posp);
  hipLaunchKernelGGL(k_pool5_fix, dim3(ngrid(NCLUST * 32, TB)), dim3(TB), 0, stream,
                     hp, posp, cnt, NCLUST);

  // ---- pooled edge attrs + basis + degree ----
  hipLaunchKernelGGL(k_fillf, dim3(1), dim3(32), 0, stream, maxab, 0.f, 1);
  hipLaunchKernelGGL(k_cart, dim3(ngrid(N_EDGES, TB)), dim3(TB), 0, stream,
                     esrc, edst, cluster, posp, cart, maxab, N_EDGES);
  hipLaunchKernelGGL(k_attr_norm, dim3(ngrid(N_EDGES * 3, TB)), dim3(TB), 0, stream,
                     cart, maxab, N_EDGES * 3);
  hipLaunchKernelGGL(k_fillf, dim3(ngrid(NCLUST, TB)), dim3(TB), 0, stream, degp, 0.f, NCLUST);
  hipLaunchKernelGGL(k_edge_basis, dim3(ngrid(N_EDGES, TB)), dim3(TB), 0, stream,
                     cart, edst, cluster, wE, degp, N_EDGES);   // wE reused for pooled graph

  // ---- cluster-level layers 6,7 (residual around both) ----
  hipLaunchKernelGGL(k_copy, dim3(ngrid(NCLUST * 32, TB)), dim3(TB), 0, stream,
                     hpsc, hp, NCLUST * 32);
  const float* pin[2]  = {hp,  hp2};
  float*       pout[2] = {hp2, hp};
  const float* res7[2] = {nullptr, hpsc};
  for (int l = 5; l < 7; ++l) {
    int Cout = coutL[l];
    hipLaunchKernelGGL(k_fillf, dim3(ngrid(NCLUST * Cout, TB)), dim3(TB), 0, stream,
                       acc, 0.f, NCLUST * Cout);
    hipLaunchKernelGGL(k_fillf, dim3(1), dim3(64), 0, stream, bnsum, 0.f, 32);
    hipLaunchKernelGGL(k_fillf, dim3(1), dim3(64), 0, stream, bnsq, 0.f, 32);
    conv_w(l);
    gemm(l, pin[l - 5], acc, cluster);
    hipLaunchKernelGGL(k_post1, dim3(ngrid(NCLUST * Cout, TB)), dim3(TB), 0, stream,
                       acc, degp, bnsum, bnsq, NCLUST, Cout);
    hipLaunchKernelGGL(k_post2, dim3(ngrid(NCLUST * Cout, TB)), dim3(TB), 0, stream,
                       acc, bnsum, bnsq, gl[l], bl[l], pout[l - 5], res7[l - 5], NCLUST, Cout);
  }

  // ---- pool7 + FC head ----
  hipLaunchKernelGGL(k_cl7, dim3(ngrid(NCLUST, TB)), dim3(TB), 0, stream, posp, cl7);
  hipLaunchKernelGGL(k_fillf, dim3(ngrid(NCELL7 * 32, TB)), dim3(TB), 0, stream,
                     hx, -INFINITY, NCELL7 * 32);
  hipLaunchKernelGGL(k_pool7, dim3(ngrid(NCLUST, TB)), dim3(TB), 0, stream, hp, cl7, hx);
  hipLaunchKernelGGL(k_hx_fix, dim3(ngrid(NCELL7 * 32, TB)), dim3(TB), 0, stream,
                     hx, NCELL7 * 32);
  hipLaunchKernelGGL(k_fc, dim3(1), dim3(32), 0, stream, hx, wfc, out);
}
